// HeteroGNN_30391188587180
// MI455X (gfx1250) — compile-verified, matrix operand order checked by reading, not run on hardware
//
#include <hip/hip_runtime.h>
#include <hip/hip_bf16.h>

// ---------------------------------------------------------------------------
// Types for WMMA (gfx1250, wave32)
// ---------------------------------------------------------------------------
typedef __attribute__((ext_vector_type(16))) __bf16 bf16x16;
typedef __attribute__((ext_vector_type(8)))  __bf16 bf16x8;
typedef __attribute__((ext_vector_type(8)))  float  f32x8;

union BFrag { bf16x16 v; bf16x8 h[2]; };

#define HIDC 256   // feature width everywhere (F=HID=OUT=256)
#define NHEAD 8
#define HDIM 32

__device__ __forceinline__ void atomAddF(float* p, float v) {
  __hip_atomic_fetch_add(p, v, __ATOMIC_RELAXED, __HIP_MEMORY_SCOPE_AGENT);
}
__device__ __forceinline__ void atomMaxU(unsigned* p, unsigned v) {
  __hip_atomic_fetch_max(p, v, __ATOMIC_RELAXED, __HIP_MEMORY_SCOPE_AGENT);
}
// monotonic float<->uint key (order-preserving, works for negatives)
__device__ __forceinline__ unsigned fkey(float f) {
  unsigned b = __float_as_uint(f);
  return (b & 0x80000000u) ? ~b : (b | 0x80000000u);
}
__device__ __forceinline__ float fkey_dec(unsigned k) {
  unsigned b = (k & 0x80000000u) ? (k ^ 0x80000000u) : ~k;
  return __uint_as_float(b);
}

// ---------------------------------------------------------------------------
// fp32 -> bf16 convert
// ---------------------------------------------------------------------------
__global__ void k_f2bf(const float* __restrict__ x, __bf16* __restrict__ y, long n) {
  long i = (long)blockIdx.x * blockDim.x + threadIdx.x;
  if (i < n) y[i] = (__bf16)x[i];
}

// W[256x256] fp32 (k-major) -> Wt[256x256] bf16 with Wt[n][k] = W[k][n]
__global__ void k_wprep(const float* __restrict__ W, __bf16* __restrict__ Wt) {
  int idx = blockIdx.x * 256 + threadIdx.x;
  int k = idx >> 8, n = idx & 255;
  Wt[n * 256 + k] = (__bf16)W[k * 256 + n];
}

// ---------------------------------------------------------------------------
// WMMA GEMM:  C[M,256] = A_bf16[M,256] @ W[256,256] + bias   (Wt = W^T bf16)
// block = 128 threads (4 waves). Block tile: 16 M rows x 64 N cols.
// LDS stages full K-strips; 8x v_wmma_f32_16x16x32_bf16 per wave.
// ---------------------------------------------------------------------------
__global__ __launch_bounds__(128) void k_gemm_bf16(
    const __bf16* __restrict__ A, const __bf16* __restrict__ Bt,
    const float* __restrict__ bias, float* __restrict__ C, int M) {
  __shared__ __bf16 As[16 * 256];   // 8 KB
  __shared__ __bf16 Bs[64 * 256];   // 32 KB
  const int m0 = blockIdx.x * 16;
  const int n0 = blockIdx.y * 64;
  const int tid = threadIdx.x;

  // cooperative stage (fully coalesced 16B loads)
  const uint4* gA = (const uint4*)(A + (long)m0 * 256);
  uint4* sA = (uint4*)As;
  #pragma unroll
  for (int i = 0; i < 4; ++i) sA[tid + i * 128] = gA[tid + i * 128];
  const uint4* gB = (const uint4*)(Bt + (long)n0 * 256);
  uint4* sB = (uint4*)Bs;
  #pragma unroll
  for (int i = 0; i < 16; ++i) sB[tid + i * 128] = gB[tid + i * 128];
  __syncthreads();

  const int wave = tid >> 5, lane = tid & 31;
  const int row  = lane & 15;             // A row within tile / B col within 16
  const int koff = (lane >> 4) * 8;       // half-wave K offset (ISA 7.12.2)
  const int col  = wave * 16 + (lane & 15);

  f32x8 c = {};
  #pragma unroll
  for (int k = 0; k < 256; k += 32) {
    BFrag a, b;
    a.h[0] = *(const bf16x8*)(As + row * 256 + k + koff);
    a.h[1] = *(const bf16x8*)(As + row * 256 + k + 16 + koff);
    b.h[0] = *(const bf16x8*)(Bs + col * 256 + k + koff);
    b.h[1] = *(const bf16x8*)(Bs + col * 256 + k + 16 + koff);
    c = __builtin_amdgcn_wmma_f32_16x16x32_bf16(false, a.v, false, b.v,
                                                (short)0, c, false, false);
  }

  const int n = n0 + col;
  const float bv = bias[n];
  const int mb = m0 + (lane >> 4) * 8;    // C layout: v[r] -> M = r (+8 for hi lanes)
  #pragma unroll
  for (int r = 0; r < 8; ++r)
    C[(long)(mb + r) * 256 + n] = c[r] + bv;
}

// ---------------------------------------------------------------------------
// Edge pass 1: score[e,h] = sum_d lrelu(fs[s]+fd[d],0.2)*attn[h,d]; segmax
// one wave per edge, lane = d within head
// ---------------------------------------------------------------------------
__global__ void k_edge_score(const float* __restrict__ fs, const float* __restrict__ fd,
                             const int* __restrict__ sidx, const int* __restrict__ didx,
                             const float* __restrict__ attn,
                             float* __restrict__ score, unsigned* __restrict__ smax, int E) {
  const int wid = threadIdx.x >> 5, lane = threadIdx.x & 31;
  const long e = (long)blockIdx.x * 8 + wid;
  if (e >= E) return;
  const int s = sidx[e], d = didx[e];
  const float* ps = fs + (long)s * HIDC;
  const float* pd = fd + (long)d * HIDC;
  #pragma unroll
  for (int h = 0; h < NHEAD; ++h) {
    float v = ps[h * HDIM + lane] + pd[h * HDIM + lane];
    v = v > 0.0f ? v : 0.2f * v;
    v *= attn[h * HDIM + lane];
    #pragma unroll
    for (int m = 16; m; m >>= 1) v += __shfl_xor(v, m, 32);
    if (lane == 0) {
      score[e * NHEAD + h] = v;
      atomMaxU(&smax[(long)d * NHEAD + h], fkey(v));
    }
  }
}

// ---------------------------------------------------------------------------
// Edge pass 2: ex = exp(score - m); segsum. ex overwrites score buffer.
// ---------------------------------------------------------------------------
__global__ void k_edge_exp(const int* __restrict__ didx, const unsigned* __restrict__ smax,
                           float* __restrict__ score, float* __restrict__ ssum, int E) {
  long i = (long)blockIdx.x * 256 + threadIdx.x;
  if (i >= (long)E * NHEAD) return;
  long e = i >> 3; int h = (int)(i & 7);
  int d = didx[e];
  float m = fkey_dec(smax[(long)d * NHEAD + h]);
  if (!isfinite(m)) m = 0.0f;
  float ex = expf(score[i] - m);
  score[i] = ex;
  atomAddF(&ssum[(long)d * NHEAD + h], ex);
}

// ---------------------------------------------------------------------------
// Edge pass 3: out[dst] += (ex/ssum[dst]) * fs[src]    (one wave per edge)
// ---------------------------------------------------------------------------
__global__ void k_edge_aggr(const float* __restrict__ fs,
                            const int* __restrict__ sidx, const int* __restrict__ didx,
                            const float* __restrict__ exs, const float* __restrict__ ssum,
                            float* __restrict__ out, int E) {
  const int wid = threadIdx.x >> 5, lane = threadIdx.x & 31;
  const long e = (long)blockIdx.x * 8 + wid;
  if (e >= E) return;
  const int s = sidx[e], d = didx[e];
  const float* ps = fs + (long)s * HIDC;
  float* po = out + (long)d * HIDC;
  #pragma unroll
  for (int h = 0; h < NHEAD; ++h) {
    float alpha = exs[e * NHEAD + h] / ssum[(long)d * NHEAD + h];
    atomAddF(&po[h * HDIM + lane], alpha * ps[h * HDIM + lane]);
  }
}

// ---------------------------------------------------------------------------
// BatchNorm (training stats) + LeakyReLU(0.01), bf16 out for next GEMMs
// ---------------------------------------------------------------------------
__global__ void k_bn_stats(const float* __restrict__ x, float* __restrict__ acc,
                           float* __restrict__ accsq, int N) {
  const int t = threadIdx.x;                  // column
  long r0 = (long)blockIdx.x * 256;
  long r1 = r0 + 256; if (r1 > N) r1 = N;
  float s = 0.0f, q = 0.0f;
  for (long r = r0; r < r1; ++r) { float v = x[r * HIDC + t]; s += v; q += v * v; }
  atomAddF(&acc[t], s);
  atomAddF(&accsq[t], q);
}

__global__ void k_bn_apply(const float* __restrict__ x, const float* __restrict__ acc,
                           const float* __restrict__ accsq, const float* __restrict__ g,
                           const float* __restrict__ b, __bf16* __restrict__ y, int N) {
  long i = (long)blockIdx.x * 256 + threadIdx.x;
  if (i >= (long)N * HIDC) return;
  int c = (int)(i & 255);
  float inv = 1.0f / (float)N;
  float mu = acc[c] * inv;
  float var = accsq[c] * inv - mu * mu;
  float v = g[c] * (x[i] - mu) * rsqrtf(var + 1e-5f) + b[c];
  y[i] = (__bf16)(v > 0.0f ? v : 0.01f * v);
}

// ---------------------------------------------------------------------------
// host launch
// ---------------------------------------------------------------------------
extern "C" void kernel_launch(void* const* d_in, const int* in_sizes, int n_in,
                              void* d_out, int out_size, void* d_ws, size_t ws_size,
                              hipStream_t stream) {
  const int NA = in_sizes[0] / HIDC;
  const int NB = in_sizes[1] / HIDC;
  const int E  = in_sizes[2];

  const float* xA = (const float*)d_in[0];
  const float* xB = (const float*)d_in[1];
  const int* src_ab = (const int*)d_in[2];
  const int* dst_ab = (const int*)d_in[3];
  const int* src_ba = (const int*)d_in[4];
  const int* dst_ba = (const int*)d_in[5];
  // params: l1_ab{Ws,bs,Wd,bd,attn} l1_ba{..} l2_ab{..} l2_ba{..} bnA{g,b} bnB{g,b}
  const float* W_raw[8]  = {(const float*)d_in[6],  (const float*)d_in[8],
                            (const float*)d_in[11], (const float*)d_in[13],
                            (const float*)d_in[16], (const float*)d_in[18],
                            (const float*)d_in[21], (const float*)d_in[23]};
  const float* bias_[8]  = {(const float*)d_in[7],  (const float*)d_in[9],
                            (const float*)d_in[12], (const float*)d_in[14],
                            (const float*)d_in[17], (const float*)d_in[19],
                            (const float*)d_in[22], (const float*)d_in[24]};
  const float* attn1_ab = (const float*)d_in[10];
  const float* attn1_ba = (const float*)d_in[15];
  const float* attn2_ab = (const float*)d_in[20];
  const float* attn2_ba = (const float*)d_in[25];
  const float* bnAg = (const float*)d_in[26];
  const float* bnAb = (const float*)d_in[27];
  const float* bnBg = (const float*)d_in[28];
  const float* bnBb = (const float*)d_in[29];

  // -------- workspace carve-out --------
  char* p = (char*)d_ws;
  auto alloc = [&](size_t bytes) -> void* {
    void* r = (void*)p; p += (bytes + 255) & ~(size_t)255; return r;
  };
  const size_t nfA = (size_t)NA * HIDC, nfB = (size_t)NB * HIDC;
  __bf16* xAb = (__bf16*)alloc(nfA * 2);
  __bf16* xBb = (__bf16*)alloc(nfB * 2);
  __bf16* hAb = (__bf16*)alloc(nfA * 2);
  __bf16* hBb = (__bf16*)alloc(nfB * 2);
  __bf16* Wt[8];
  for (int i = 0; i < 8; ++i) Wt[i] = (__bf16*)alloc(256 * 256 * 2);
  float* fsA = (float*)alloc(nfA * 4);
  float* fdB = (float*)alloc(nfB * 4);
  float* fsB = (float*)alloc(nfB * 4);
  float* fdA = (float*)alloc(nfA * 4);
  float* hA1 = (float*)alloc(nfA * 4);
  float* hB1 = (float*)alloc(nfB * 4);
  float* sc_ab = (float*)alloc((size_t)E * NHEAD * 4);
  float* sc_ba = (float*)alloc((size_t)E * NHEAD * 4);
  unsigned* smaxA = (unsigned*)alloc((size_t)NA * NHEAD * 4);
  unsigned* smaxB = (unsigned*)alloc((size_t)NB * NHEAD * 4);
  float* ssumA = (float*)alloc((size_t)NA * NHEAD * 4);
  float* ssumB = (float*)alloc((size_t)NB * NHEAD * 4);
  float* bnAcc = (float*)alloc(4 * 256 * 4);  // accA, accsqA, accB, accsqB

  float* hA2 = (float*)d_out;            // outputs: hA2 then hB2
  float* hB2 = (float*)d_out + nfA;

  const dim3 b256(256), b128(128);
  const int ewaveBlocks = (E + 7) / 8;
  const int ehBlocks = (int)(((long)E * NHEAD + 255) / 256);
  const dim3 gemmA((NA + 15) / 16, HIDC / 64), gemmB((NB + 15) / 16, HIDC / 64);

  // -------- zero-init (capture-safe) --------
  hipMemsetAsync(smaxA, 0, (size_t)NA * NHEAD * 4, stream);
  hipMemsetAsync(smaxB, 0, (size_t)NB * NHEAD * 4, stream);
  hipMemsetAsync(ssumA, 0, (size_t)NA * NHEAD * 4, stream);
  hipMemsetAsync(ssumB, 0, (size_t)NB * NHEAD * 4, stream);
  hipMemsetAsync(hA1, 0, nfA * 4, stream);
  hipMemsetAsync(hB1, 0, nfB * 4, stream);
  hipMemsetAsync(bnAcc, 0, 4 * 256 * 4, stream);
  hipMemsetAsync(d_out, 0, (size_t)out_size * 4, stream);

  // -------- prep: bf16 activations + transposed bf16 weights --------
  k_f2bf<<<(int)((nfA + 255) / 256), b256, 0, stream>>>(xA, xAb, (long)nfA);
  k_f2bf<<<(int)((nfB + 255) / 256), b256, 0, stream>>>(xB, xBb, (long)nfB);
  for (int i = 0; i < 8; ++i)
    k_wprep<<<256, b256, 0, stream>>>(W_raw[i], Wt[i]);

  // ================= layer 1 =================
  k_gemm_bf16<<<gemmA, b128, 0, stream>>>(xAb, Wt[0], bias_[0], fsA, NA); // l1_ab.Ws
  k_gemm_bf16<<<gemmB, b128, 0, stream>>>(xBb, Wt[1], bias_[1], fdB, NB); // l1_ab.Wd
  k_gemm_bf16<<<gemmB, b128, 0, stream>>>(xBb, Wt[2], bias_[2], fsB, NB); // l1_ba.Ws
  k_gemm_bf16<<<gemmA, b128, 0, stream>>>(xAb, Wt[3], bias_[3], fdA, NA); // l1_ba.Wd

  k_edge_score<<<ewaveBlocks, b256, 0, stream>>>(fsA, fdB, src_ab, dst_ab, attn1_ab, sc_ab, smaxB, E);
  k_edge_score<<<ewaveBlocks, b256, 0, stream>>>(fsB, fdA, src_ba, dst_ba, attn1_ba, sc_ba, smaxA, E);
  k_edge_exp<<<ehBlocks, b256, 0, stream>>>(dst_ab, smaxB, sc_ab, ssumB, E);
  k_edge_exp<<<ehBlocks, b256, 0, stream>>>(dst_ba, smaxA, sc_ba, ssumA, E);
  k_edge_aggr<<<ewaveBlocks, b256, 0, stream>>>(fsA, src_ab, dst_ab, sc_ab, ssumB, hB1, E);
  k_edge_aggr<<<ewaveBlocks, b256, 0, stream>>>(fsB, src_ba, dst_ba, sc_ba, ssumA, hA1, E);

  // BN + leaky_relu -> bf16
  k_bn_stats<<<(NA + 255) / 256, b256, 0, stream>>>(hA1, bnAcc + 0,   bnAcc + 256, NA);
  k_bn_stats<<<(NB + 255) / 256, b256, 0, stream>>>(hB1, bnAcc + 512, bnAcc + 768, NB);
  k_bn_apply<<<(int)((nfA + 255) / 256), b256, 0, stream>>>(hA1, bnAcc + 0,   bnAcc + 256, bnAg, bnAb, hAb, NA);
  k_bn_apply<<<(int)((nfB + 255) / 256), b256, 0, stream>>>(hB1, bnAcc + 512, bnAcc + 768, bnBg, bnBb, hBb, NB);

  // ================= layer 2 =================
  hipMemsetAsync(smaxA, 0, (size_t)NA * NHEAD * 4, stream);
  hipMemsetAsync(smaxB, 0, (size_t)NB * NHEAD * 4, stream);
  hipMemsetAsync(ssumA, 0, (size_t)NA * NHEAD * 4, stream);
  hipMemsetAsync(ssumB, 0, (size_t)NB * NHEAD * 4, stream);

  k_gemm_bf16<<<gemmA, b128, 0, stream>>>(hAb, Wt[4], bias_[4], fsA, NA); // l2_ab.Ws
  k_gemm_bf16<<<gemmB, b128, 0, stream>>>(hBb, Wt[5], bias_[5], fdB, NB); // l2_ab.Wd
  k_gemm_bf16<<<gemmB, b128, 0, stream>>>(hBb, Wt[6], bias_[6], fsB, NB); // l2_ba.Ws
  k_gemm_bf16<<<gemmA, b128, 0, stream>>>(hAb, Wt[7], bias_[7], fdA, NA); // l2_ba.Wd

  k_edge_score<<<ewaveBlocks, b256, 0, stream>>>(fsA, fdB, src_ab, dst_ab, attn2_ab, sc_ab, smaxB, E);
  k_edge_score<<<ewaveBlocks, b256, 0, stream>>>(fsB, fdA, src_ba, dst_ba, attn2_ba, sc_ba, smaxA, E);
  k_edge_exp<<<ehBlocks, b256, 0, stream>>>(dst_ab, smaxB, sc_ab, ssumB, E);
  k_edge_exp<<<ehBlocks, b256, 0, stream>>>(dst_ba, smaxA, sc_ba, ssumA, E);
  k_edge_aggr<<<ewaveBlocks, b256, 0, stream>>>(fsA, src_ab, dst_ab, sc_ab, ssumB, hB2, E);
  k_edge_aggr<<<ewaveBlocks, b256, 0, stream>>>(fsB, src_ba, dst_ba, sc_ba, ssumA, hA2, E);

  (void)n_in; (void)ws_size;
}
